// ConvFlow_2216203125311
// MI455X (gfx1250) — matrix operand order repaired and checked by reading, not compiled, not run on hardware
//
#include <hip/hip_runtime.h>
#include <math.h>

#define HID   192
#define TILE  128
#define HALO  16
#define WTOT  (TILE + 2*HALO)   // 160 positions per tile (incl. halo)
#define XS    WTOT              // residual stream stride (floats)
#define YS    200               // bf16 activation stride (padded)
#define ZS    196               // fp32 matmul-out stride (padded)
#define TLEN  8192
#define NB    32
#define NBINS 10
#define TAILF 5.0f

typedef __attribute__((ext_vector_type(16))) __bf16 v16bf;
typedef __attribute__((ext_vector_type(8)))  float  v8f;

union ABf { unsigned int u[8]; v16bf v; };

__device__ __forceinline__ float sp_softplus(float x) {
  return (x > 20.f) ? x : log1pf(expf(x));
}
__device__ __forceinline__ float gelu_exact(float x) {
  return 0.5f * x * (1.f + erff(x * 0.70710678118654752f));
}

// Convert fp32 weights to bf16 once: pw_w (3*192*192) then proj_w padded to 32 rows.
__global__ void prep_weights(const float* __restrict__ pw_w,
                             const float* __restrict__ proj_w,
                             unsigned short* __restrict__ wsb) {
  int i = blockIdx.x * blockDim.x + threadIdx.x;
  const int NPW = 3 * HID * HID;
  const int NPJ = 32 * HID;
  if (i < NPW) {
    union { __bf16 h; unsigned short s; } u;
    u.h = (__bf16)pw_w[i];
    wsb[i] = u.s;
  } else if (i < NPW + NPJ) {
    int j = i - NPW;
    int r = j / HID, c = j - r * HID;
    float v = (r < 29) ? proj_w[r * HID + c] : 0.f;
    union { __bf16 h; unsigned short s; } u;
    u.h = (__bf16)v;
    wsb[i] = u.s;
  }
}

__global__ void zero_logdet(float* __restrict__ p) {
  if (threadIdx.x < NB) p[threadIdx.x] = 0.f;
}

__global__ __launch_bounds__(256, 1)
void convflow_fused(const float* __restrict__ x, const float* __restrict__ xmask,
                    const float* __restrict__ in_w, const float* __restrict__ in_b,
                    const float* __restrict__ dw_w, const float* __restrict__ dw_b,
                    const float* __restrict__ ln1_g, const float* __restrict__ ln1_b,
                    const float* __restrict__ pw_b,
                    const float* __restrict__ ln2_g, const float* __restrict__ ln2_b,
                    const float* __restrict__ proj_b,
                    const unsigned short* __restrict__ wsb,
                    float* __restrict__ out) {
  extern __shared__ char smem[];
  float*  Xs    = (float*)smem;                                         // [HID][XS]
  __bf16* Ys    = (__bf16*)(smem + (size_t)HID*XS*4);                   // [WTOT][YS]
  float*  Zs    = (float*)(smem + (size_t)HID*XS*4 + (size_t)WTOT*YS*2);// [WTOT][ZS]
  float*  maskS = (float*)(smem + (size_t)HID*XS*4 + (size_t)WTOT*YS*2
                                + (size_t)WTOT*ZS*4);                   // [WTOT]
  __shared__ float red[256];

  const int tid  = threadIdx.x;
  const int b    = blockIdx.y;
  const int t0   = blockIdx.x * TILE;
  const int wave = tid >> 5, lane = tid & 31;
  const int row  = lane & 15, hi = lane >> 4;

  const __bf16* pwB = (const __bf16*)wsb;
  const __bf16* pjB = pwB + 3 * HID * HID;

  // mask tile (0 outside [0,T) replicates conv zero-padding)
  for (int w = tid; w < WTOT; w += 256) {
    int t = t0 - HALO + w;
    maskS[w] = (t >= 0 && t < TLEN) ? xmask[b * TLEN + t] : 0.f;
  }
  // h0 = in_w * xa + in_b  (1x1 conv, CIN half = 1 channel)
  for (int i = tid; i < HID * WTOT; i += 256) {
    int c = i / WTOT, w = i - c * WTOT;
    int t = t0 - HALO + w;
    float xv = (t >= 0 && t < TLEN) ? x[(b * 2 + 0) * TLEN + t] : 0.f;
    Xs[c * XS + w] = in_w[c] * xv + in_b[c];
  }
  __syncthreads();

  int dil = 1;
  for (int layer = 0; layer < 3; ++layer) {
    const float* dwW = dw_w + layer * HID * 3;
    const float* dwB = dw_b + layer * HID;
    const float* g1  = ln1_g + layer * HID;
    const float* b1  = ln1_b + layer * HID;
    // Phase A: masked dilated depthwise conv -> LN(channel) -> GELU -> bf16
    for (int w = tid; w < WTOT; w += 256) {
      float mm = (w >= dil)        ? maskS[w - dil] : 0.f;
      float mc = maskS[w];
      float mp = (w + dil < WTOT)  ? maskS[w + dil] : 0.f;
      float s1 = 0.f, s2 = 0.f;
      for (int c = 0; c < HID; ++c) {
        float xm = (w >= dil)       ? Xs[c * XS + w - dil] * mm : 0.f;
        float x0 = Xs[c * XS + w] * mc;
        float xp = (w + dil < WTOT) ? Xs[c * XS + w + dil] * mp : 0.f;
        float acc = dwB[c] + dwW[c*3+0]*xm + dwW[c*3+1]*x0 + dwW[c*3+2]*xp;
        Zs[w * ZS + c] = acc;
        s1 += acc; s2 += acc * acc;
      }
      float mu  = s1 * (1.f / HID);
      float var = s2 * (1.f / HID) - mu * mu;
      float inv = rsqrtf(var + 1e-5f);
      for (int c = 0; c < HID; ++c) {
        float v = (Zs[w * ZS + c] - mu) * inv * g1[c] + b1[c];
        Ys[w * YS + c] = (__bf16)gelu_exact(v);
      }
    }
    __syncthreads();
    // Phase B: 192x192 pointwise matmul via WMMA bf16.
    // Register-blocked: each wave owns 1 M-tile x 5 N-tiles (one A load feeds
    // 5 independent WMMA accumulator chains -> hides XDL latency, 5x fewer
    // global weight loads per WMMA). 12 M-tiles x 2 N-groups = 24 groups.
    const __bf16* Wl = pwB + layer * HID * HID;
    const float*  pb = pw_b + layer * HID;
    for (int grp = wave; grp < 24; grp += 8) {
      int mt = grp % 12, ng = grp / 12;
      int m0 = mt * 16, wbase = ng * 80;
      v8f acc[5] = {};
      for (int kt = 0; kt < 6; ++kt) {
        int c0 = kt * 32;
        ABf A;
        const unsigned int* ar0 =
            (const unsigned int*)(Wl + (m0 + row) * HID + c0 + hi * 8);
        const unsigned int* ar1 =
            (const unsigned int*)(Wl + (m0 + row) * HID + c0 + 16 + hi * 8);
        #pragma unroll
        for (int j = 0; j < 4; ++j) { A.u[j] = ar0[j]; A.u[4 + j] = ar1[j]; }
        #pragma unroll
        for (int n = 0; n < 5; ++n) {
          ABf Bv;
          const __bf16* br = Ys + (wbase + n * 16 + row) * YS + c0 + hi * 8;
          #pragma unroll
          for (int j = 0; j < 8; ++j) Bv.u[j] = *(const unsigned int*)(br + 2 * j + (j >= 4 ? 8 : 0));
          acc[n] = __builtin_amdgcn_wmma_f32_16x16x32_bf16(
              false, A.v, false, Bv.v, (short)0, acc[n], false, false);
        }
      }
      int cD = m0 + hi * 8;
      #pragma unroll
      for (int n = 0; n < 5; ++n) {
        int wD = wbase + n * 16 + row;
        #pragma unroll
        for (int r = 0; r < 8; ++r) Zs[wD * ZS + cD + r] = acc[n][r] + pb[cD + r];
      }
    }
    __syncthreads();
    // Phase C: LN -> GELU -> residual add into X
    const float* g2 = ln2_g + layer * HID;
    const float* b2 = ln2_b + layer * HID;
    for (int w = tid; w < WTOT; w += 256) {
      float s1 = 0.f, s2 = 0.f;
      for (int c = 0; c < HID; ++c) { float v = Zs[w * ZS + c]; s1 += v; s2 += v * v; }
      float mu  = s1 * (1.f / HID);
      float var = s2 * (1.f / HID) - mu * mu;
      float inv = rsqrtf(var + 1e-5f);
      for (int c = 0; c < HID; ++c) {
        float v = (Zs[w * ZS + c] - mu) * inv * g2[c] + b2[c];
        Xs[c * XS + w] += gelu_exact(v);
      }
    }
    __syncthreads();
    dil *= 3;
  }

  // Masked residual stream -> bf16 for projection matmul
  for (int w = tid; w < WTOT; w += 256) {
    float m = maskS[w];
    for (int c = 0; c < HID; ++c) Ys[w * YS + c] = (__bf16)(Xs[c * XS + w] * m);
  }
  __syncthreads();
  // Projection: (29 padded to 32) x 192, central 128 positions.
  // 2 M-tiles x 2 N-groups (4 tiles each) = 4 wave-groups.
  for (int grp = wave; grp < 4; grp += 8) {
    int mt = grp & 1, ng = grp >> 1;
    int m0 = mt * 16, wbase = HALO + ng * 64;
    v8f acc[4] = {};
    for (int kt = 0; kt < 6; ++kt) {
      int c0 = kt * 32;
      ABf A;
      const unsigned int* ar0 =
          (const unsigned int*)(pjB + (m0 + row) * HID + c0 + hi * 8);
      const unsigned int* ar1 =
          (const unsigned int*)(pjB + (m0 + row) * HID + c0 + 16 + hi * 8);
      #pragma unroll
      for (int j = 0; j < 4; ++j) { A.u[j] = ar0[j]; A.u[4 + j] = ar1[j]; }
      #pragma unroll
      for (int n = 0; n < 4; ++n) {
        ABf Bv;
        const __bf16* br = Ys + (wbase + n * 16 + row) * YS + c0 + hi * 8;
        #pragma unroll
        for (int j = 0; j < 8; ++j) Bv.u[j] = *(const unsigned int*)(br + 2 * j + (j >= 4 ? 8 : 0));
        acc[n] = __builtin_amdgcn_wmma_f32_16x16x32_bf16(
            false, A.v, false, Bv.v, (short)0, acc[n], false, false);
      }
    }
    int cD = m0 + hi * 8;
    #pragma unroll
    for (int n = 0; n < 4; ++n) {
      int wD = wbase + n * 16 + row;
      #pragma unroll
      for (int r = 0; r < 8; ++r) Zs[wD * ZS + cD + r] = acc[n][r];
    }
  }
  __syncthreads();

  // Rational-quadratic spline per central position
  float ladv = 0.f;
  if (tid < TILE) {
    int w = HALO + tid;
    int t = t0 + tid;
    float m = maskS[w];
    float pr[29];
    for (int p = 0; p < 29; ++p) pr[p] = (Zs[w * ZS + p] + proj_b[p]) * m;
    const float inv_dn = 0.07216878364870322f; // 1/sqrt(192)

    float cw[NBINS + 1], wd[NBINS], ch[NBINS + 1], hg[NBINS], der[NBINS + 1];
    { // widths
      float mx = -3.4e38f;
      for (int k = 0; k < NBINS; ++k) mx = fmaxf(mx, pr[k] * inv_dn);
      float ex[NBINS], s = 0.f;
      for (int k = 0; k < NBINS; ++k) { ex[k] = expf(pr[k] * inv_dn - mx); s += ex[k]; }
      float run = 0.f; cw[0] = -TAILF;
      for (int k = 0; k < NBINS; ++k) {
        float wk = 0.001f + (1.f - 0.001f * NBINS) * (ex[k] / s);
        run += wk;
        cw[k + 1] = 2.f * TAILF * run - TAILF;
      }
      cw[NBINS] = TAILF;
      for (int k = 0; k < NBINS; ++k) wd[k] = cw[k + 1] - cw[k];
    }
    { // heights
      float mx = -3.4e38f;
      for (int k = 0; k < NBINS; ++k) mx = fmaxf(mx, pr[10 + k] * inv_dn);
      float ex[NBINS], s = 0.f;
      for (int k = 0; k < NBINS; ++k) { ex[k] = expf(pr[10 + k] * inv_dn - mx); s += ex[k]; }
      float run = 0.f; ch[0] = -TAILF;
      for (int k = 0; k < NBINS; ++k) {
        float hk = 0.001f + (1.f - 0.001f * NBINS) * (ex[k] / s);
        run += hk;
        ch[k + 1] = 2.f * TAILF * run - TAILF;
      }
      ch[NBINS] = TAILF;
      for (int k = 0; k < NBINS; ++k) hg[k] = ch[k + 1] - ch[k];
    }
    { // derivatives (padded with const at both ends)
      const float cpad = logf(expf(0.999f) - 1.f);
      der[0] = 0.001f + sp_softplus(cpad);
      for (int k = 0; k < 9; ++k) der[k + 1] = 0.001f + sp_softplus(pr[20 + k]);
      der[NBINS] = 0.001f + sp_softplus(cpad);
    }

    float xin = x[(b * 2 + 1) * TLEN + t];
    int inside = (xin >= -TAILF) && (xin <= TAILF);
    float xc = fminf(fmaxf(xin, -TAILF), TAILF);
    int idx = 0;
    for (int k = 1; k < NBINS; ++k) if (xc >= cw[k]) idx = k;

    float icw = cw[idx], ibw = wd[idx], ich = ch[idx], ih = hg[idx];
    float idl = ih / ibw;
    float dk = der[idx], dk1 = der[idx + 1];
    float th = (xc - icw) / ibw;
    float t1m = th * (1.f - th);
    float num = ih * (idl * th * th + dk * t1m);
    float den = idl + (dk + dk1 - 2.f * idl) * t1m;
    float outv = ich + num / den;
    float omth = 1.f - th;
    float dnum = idl * idl * (dk1 * th * th + 2.f * idl * t1m + dk * omth * omth);
    float lad = logf(dnum) - 2.f * logf(den);

    float xb2 = inside ? outv : xin;
    ladv = (inside ? lad : 0.f) * m;
    float xav = x[(b * 2 + 0) * TLEN + t];
    out[(b * 2 + 0) * TLEN + t] = xav * m;
    out[(b * 2 + 1) * TLEN + t] = xb2 * m;
  }
  red[tid] = ladv;
  __syncthreads();
  for (int s = 128; s > 0; s >>= 1) {
    if (tid < s) red[tid] += red[tid + s];
    __syncthreads();
  }
  if (tid == 0) atomicAdd(out + NB * 2 * TLEN + b, red[0]);
}

extern "C" void kernel_launch(void* const* d_in, const int* in_sizes, int n_in,
                              void* d_out, int out_size, void* d_ws, size_t ws_size,
                              hipStream_t stream) {
  (void)in_sizes; (void)n_in; (void)out_size; (void)ws_size;
  const float* x      = (const float*)d_in[0];
  const float* xmask  = (const float*)d_in[1];
  const float* in_w   = (const float*)d_in[2];
  const float* in_b   = (const float*)d_in[3];
  const float* dw_w   = (const float*)d_in[4];
  const float* dw_b   = (const float*)d_in[5];
  const float* ln1_g  = (const float*)d_in[6];
  const float* ln1_b  = (const float*)d_in[7];
  const float* pw_w   = (const float*)d_in[8];
  const float* pw_b   = (const float*)d_in[9];
  const float* ln2_g  = (const float*)d_in[10];
  const float* ln2_b  = (const float*)d_in[11];
  const float* proj_w = (const float*)d_in[12];
  const float* proj_b = (const float*)d_in[13];
  float* out = (float*)d_out;
  unsigned short* wsb = (unsigned short*)d_ws;

  int nconv = 3 * HID * HID + 32 * HID;
  prep_weights<<<(nconv + 255) / 256, 256, 0, stream>>>(pw_w, proj_w, wsb);
  zero_logdet<<<1, 64, 0, stream>>>(out + NB * 2 * TLEN);

  size_t smem = (size_t)HID * XS * 4 + (size_t)WTOT * YS * 2
              + (size_t)WTOT * ZS * 4 + (size_t)WTOT * 4;
  (void)hipFuncSetAttribute((const void*)convflow_fused,
                            hipFuncAttributeMaxDynamicSharedMemorySize, (int)smem);
  convflow_fused<<<dim3(TLEN / TILE, NB), 256, smem, stream>>>(
      x, xmask, in_w, in_b, dw_w, dw_b, ln1_g, ln1_b, pw_b, ln2_g, ln2_b,
      proj_b, wsb, out);
}
